// ObservationEncoder_81054622810353
// MI455X (gfx1250) — matrix operand order repaired and checked by reading, not compiled, not run on hardware
//
#include <hip/hip_runtime.h>
#include <hip/hip_bf16.h>
#include <cstdint>
#include <cstddef>

typedef _Float16 half_t;
typedef __attribute__((ext_vector_type(16))) _Float16 v16h;
typedef __attribute__((ext_vector_type(8)))  _Float16 v8h;
typedef __attribute__((ext_vector_type(8)))  float    v8f;
typedef __attribute__((ext_vector_type(4)))  unsigned int u32x4;
typedef __attribute__((ext_vector_type(8)))  unsigned int u32x8;

// ---------------------------------------------------------------------------
// weight repack: f32 (Cout,Ktot) -> f16 (Cout,Kpad) zero-padded rows
// ---------------------------------------------------------------------------
__global__ void repack_w_kernel(const float* __restrict__ src, half_t* __restrict__ dst,
                                int Cout, int Ktot, int Kpad) {
    int i = blockIdx.x * blockDim.x + threadIdx.x;
    if (i < Cout * Kpad) {
        int co = i / Kpad;
        int k = i - co * Kpad;
        dst[i] = (k < Ktot) ? (half_t)src[(size_t)co * Ktot + k] : (half_t)0.0f;
    }
}

// ---------------------------------------------------------------------------
// any_nz: one block per (b,o) pair; OR-reduce seg[b,o,:,:,:] != 0
// ---------------------------------------------------------------------------
__global__ void anynz_kernel(const float* __restrict__ seg, float* __restrict__ anynz) {
    const int P = 4 * 128 * 128;                  // FRM*IMG*IMG
    const float* s = seg + (size_t)blockIdx.x * P;
    int found = 0;
    for (int i = threadIdx.x; i < P; i += blockDim.x) found |= (s[i] != 0.0f);
    __shared__ int sh[256];
    sh[threadIdx.x] = found;
    __syncthreads();
    for (int st = 128; st > 0; st >>= 1) {
        if ((int)threadIdx.x < st) sh[threadIdx.x] |= sh[threadIdx.x + st];
        __syncthreads();
    }
    if (threadIdx.x == 0) anynz[blockIdx.x] = sh[0] ? 1.0f : 0.0f;
}

// ---------------------------------------------------------------------------
// preprocess: x[n=b*8+o, c, h, w] (128,6,128,128) f16
// ---------------------------------------------------------------------------
__global__ void preprocess_kernel(const float* __restrict__ obs,
                                  const float* __restrict__ seg,
                                  const float* __restrict__ anynz,
                                  half_t* __restrict__ x) {
    const int TOT = 128 * 6 * 128 * 128;
    for (int idx = blockIdx.x * blockDim.x + threadIdx.x; idx < TOT;
         idx += gridDim.x * blockDim.x) {
        int w = idx & 127;
        int h = (idx >> 7) & 127;
        int c = (idx >> 14) % 6;
        int n = (idx >> 14) / 6;
        int b = n >> 3, o = n & 7;
        float acc = 0.0f;
        if (c < 3) {
            for (int f = 0; f < 4; ++f) {
                float ov = obs[((((b * 4 + f) * 128 + h) * 128) + w) * 3 + c];
                float sv = seg[((((b * 8 + o) * 4 + f) * 128) + h) * 128 + w];
                acc += ov * sv;
            }
        } else {
            float s = 0.0f;
            for (int f = 0; f < 4; ++f)
                s += obs[((((b * 4 + f) * 128 + h) * 128) + w) * 3 + (c - 3)];
            acc = s * anynz[b * 8 + o];
        }
        x[idx] = (half_t)acc;
    }
}

// ---------------------------------------------------------------------------
// Implicit-GEMM conv via v_wmma_f32_16x16x32_f16.
// Block = 8 waves sharing one 16-Cout stripe (tm); 512 pixels per block.
// Weight slab (16x32 f16, row stride Kpad) is DMA'd per K-step by the
// Tensor Data Mover (tensor_load_to_lds, TENSORcnt) into a double-buffered
// LDS slab; A-fragments are two aligned ds b128 reads.
// ---------------------------------------------------------------------------
template <int KHW, int S, int P>
__global__ __launch_bounds__(256)
void conv_wmma_kernel(const half_t* __restrict__ in, const half_t* __restrict__ wt,
                      float* __restrict__ out,
                      int Nimg, int Cin, int H, int W, int Cout, int OH, int OW) {
    constexpr int NT = 4;                        // 4 pixel subtiles per wave
    const int Ktot = Cin * KHW * KHW;
    const int Kpad = (Ktot + 31) & ~31;
    const int OHW = OH * OW;
    const int Npix = Nimg * OHW;
    const int tiles_m = Cout >> 4;
    const int tm = blockIdx.x % tiles_m;
    const int pixblk = (blockIdx.x / tiles_m) * 512;
    const int tid = threadIdx.x;
    const int wid = tid >> 5;
    const int lane = tid & 31;
    const int hi = lane >> 4;
    const int lm = lane & 15;

    __shared__ __align__(16) half_t slab[2][16 * 32];   // double-buffered A slab

    // per-subtile pixel decode (fixed for whole kernel); pbase2 in BYTES
    int pbase2[NT], pih0[NT], piw0[NT], prem[NT], pnn[NT];
    bool pv[NT];
#pragma unroll
    for (int t = 0; t < NT; ++t) {
        int pix = pixblk + wid * 64 + t * 16 + lm;
        pv[t] = pix < Npix;
        int pc = pv[t] ? pix : 0;
        int pn = pc / OHW;
        int rem = pc - pn * OHW;
        int oh = rem / OW, ow = rem - (rem / OW) * OW;
        pih0[t] = oh * S - P;
        piw0[t] = ow * S - P;
        pbase2[t] = (pn * Cin * H * W + pih0[t] * W + piw0[t]) * 2;
        pnn[t] = pn;
        prem[t] = rem;
    }

    // ---- TDM descriptor: group1 is loop-invariant ----
    u32x8 g1;
    g1[0] = 0x00010000u;                                   // mask=0, data_size=2B
    g1[1] = ((unsigned)Kpad & 0xFFFFu) << 16;              // tensor_dim0 lo16 @bit48
    g1[2] = ((unsigned)Kpad >> 16) |
            (((unsigned)Cout & 0xFFFFu) << 16);            // dim0 hi16, dim1 lo16
    g1[3] = (((unsigned)Cout >> 16) & 0xFFFFu) |
            (32u << 16);                                   // dim1 hi16, tile_dim0=32
    g1[4] = 16u;                                           // tile_dim1=16, tile_dim2=0
    g1[5] = (unsigned)Kpad;                                // tensor_dim0_stride lo32
    g1[6] = 0u;                                            // stride hi16, dim1_stride lo16
    g1[7] = 0u;

    const uint32_t lds_base = (uint32_t)(uintptr_t)(&slab[0][0]);
    const half_t* wbase = wt + (size_t)tm * 16 * Kpad;

    auto issue = [&](int k0, int buf) {
        if (wid == 0) {                                    // wave-uniform: one TDM issue
            uint64_t ga = (uint64_t)(uintptr_t)(wbase + k0);
            u32x4 g0;
            g0[0] = 1u;                                    // count=1 (valid descriptor)
            g0[1] = lds_base + (unsigned)buf * 1024u;      // lds_addr (bytes)
            g0[2] = (unsigned)ga;                          // global_addr[31:0]
            g0[3] = (unsigned)(ga >> 32) | (2u << 30);     // global_addr[56:32] | type=2
            asm volatile("tensor_load_to_lds %0, %1" :: "s"(g0), "s"(g1) : "memory");
        }
    };

    v8f acc[NT];
#pragma unroll
    for (int t = 0; t < NT; ++t)
#pragma unroll
        for (int i = 0; i < 8; ++i) acc[t][i] = 0.0f;

    const int nsteps = Kpad >> 5;
    issue(0, 0);
    for (int s = 0; s < nsteps; ++s) {
        if (wid == 0) __builtin_amdgcn_s_wait_tensorcnt(0);
        __syncthreads();
        if (s + 1 < nsteps) issue((s + 1) * 32, (s + 1) & 1);
        const int buf = s & 1;

        // A fragment from LDS: two aligned b128 reads
        const v8h* rowp = (const v8h*)&slab[buf][lm * 32];
        v8h alo = rowp[2 * hi];
        v8h ahi = rowp[2 * hi + 1];
        v16h a = __builtin_shufflevector(alo, ahi, 0, 1, 2, 3, 4, 5, 6, 7,
                                         8, 9, 10, 11, 12, 13, 14, 15);

        // B fragments: division-free carry-incremented im2col gather (byte offs)
        const int kk = s * 32 + 16 * hi;
        int cin = kk / (KHW * KHW);
        int rr = kk - cin * (KHW * KHW);
        int kh = rr / KHW;
        int kw = rr - kh * KHW;
        int koff = ((cin * H + kh) * W + kw) * 2;
        const char* inB = (const char*)in;
        v16h bf[NT];
#pragma unroll
        for (int e = 0; e < 16; ++e) {
            const bool kvalid = (kk + e) < Ktot;
#pragma unroll
            for (int t = 0; t < NT; ++t) {
                int ih = pih0[t] + kh, iw = piw0[t] + kw;
                bool valid = kvalid && pv[t] &&
                             ((unsigned)ih < (unsigned)H) &&
                             ((unsigned)iw < (unsigned)W);
                int a2 = valid ? (pbase2[t] + koff) : 0;
                half_t vv = *(const half_t*)(inB + a2);
                bf[t][e] = valid ? vv : (half_t)0.0f;
            }
            ++kw; koff += 2;
            if (kw == KHW) {
                kw = 0; ++kh; koff += 2 * (W - KHW);
                if (kh == KHW) { kh = 0; ++cin; koff += 2 * (H - KHW) * W; }
            }
        }
#pragma unroll
        for (int t = 0; t < NT; ++t)
            acc[t] = __builtin_amdgcn_wmma_f32_16x16x32_f16(
                false, a, false, bf[t], (short)0, acc[t], false, false);
        __syncthreads();
    }

    // D store: VGPR v -> M = v + 8*hi (Cout), N = lm (pixel)
#pragma unroll
    for (int t = 0; t < NT; ++t) {
        if (!pv[t]) continue;
        int base = pnn[t] * Cout * OHW + prem[t];
#pragma unroll
        for (int v = 0; v < 8; ++v) {
            int cout = tm * 16 + v + 8 * hi;
            out[base + cout * OHW] = acc[t][v];
        }
    }
}

// ---------------------------------------------------------------------------
// BN statistics: one block per channel; division-free loops
// ---------------------------------------------------------------------------
__global__ void bn_stats_kernel(const float* __restrict__ x, int Nimg, int C, int HW,
                                float* __restrict__ mean, float* __restrict__ rstd,
                                float eps) {
    int c = blockIdx.x;
    float s = 0.0f, s2 = 0.0f;
    for (int n = 0; n < Nimg; ++n) {
        const float* p = x + ((size_t)n * C + c) * HW;
        for (int r = threadIdx.x; r < HW; r += blockDim.x) {
            float v = p[r];
            s += v;
            s2 += v * v;
        }
    }
    __shared__ float ss[256], sq[256];
    ss[threadIdx.x] = s;
    sq[threadIdx.x] = s2;
    __syncthreads();
    for (int st = 128; st > 0; st >>= 1) {
        if ((int)threadIdx.x < st) {
            ss[threadIdx.x] += ss[threadIdx.x + st];
            sq[threadIdx.x] += sq[threadIdx.x + st];
        }
        __syncthreads();
    }
    if (threadIdx.x == 0) {
        float cnt = (float)Nimg * (float)HW;
        float m = ss[0] / cnt;
        float var = sq[0] / cnt - m * m;
        mean[c] = m;
        rstd[c] = rsqrtf(var + eps);
    }
}

// ---------------------------------------------------------------------------
// BN apply + optional residual + optional ReLU; 2-D grid: blockIdx.y = plane
// ---------------------------------------------------------------------------
__global__ void bn_apply_kernel(const float* x, const float* mean, const float* rstd,
                                const float* g, const float* b, int C, int HW,
                                const float* residF, const half_t* residH, int relu,
                                half_t* outH, float* outF) {
    int plane = blockIdx.y;                       // n*C + c
    int c = plane % C;
    float mu = mean[c], rs = rstd[c], gg = g[c], bb = b[c];
    size_t base = (size_t)plane * HW;
    for (int r = blockIdx.x * blockDim.x + threadIdx.x; r < HW;
         r += gridDim.x * blockDim.x) {
        size_t idx = base + r;
        float y = (x[idx] - mu) * rs * gg + bb;
        if (residF) y += residF[idx];
        if (residH) y += (float)residH[idx];
        if (relu) y = fmaxf(y, 0.0f);
        if (outH) outH[idx] = (half_t)y;
        if (outF) outF[idx] = y;
    }
}

// ---------------------------------------------------------------------------
// stem maxpool, specialized: (128,64,64,64) -> (128,64,32,32), 3x3 s2 p1
// ---------------------------------------------------------------------------
__global__ void maxpool_kernel(const half_t* __restrict__ in, half_t* __restrict__ outH,
                               float* __restrict__ outF) {
    const int TOT = 128 * 64 * 32 * 32;
    int idx = blockIdx.x * blockDim.x + threadIdx.x;
    if (idx >= TOT) return;
    int ow = idx & 31;
    int oh = (idx >> 5) & 31;
    int nc = idx >> 10;
    const half_t* p = in + (size_t)nc * 4096;
    float m = -INFINITY;
    for (int kh = 0; kh < 3; ++kh) {
        int ih = oh * 2 - 1 + kh;
        if ((unsigned)ih >= 64u) continue;
        for (int kw = 0; kw < 3; ++kw) {
            int iw = ow * 2 - 1 + kw;
            if ((unsigned)iw >= 64u) continue;
            m = fmaxf(m, (float)p[ih * 64 + iw]);
        }
    }
    outH[idx] = (half_t)m;
    outF[idx] = m;
}

// ---------------------------------------------------------------------------
// global average pool, specialized: (128,512,4,4) -> (128,512) f16
// ---------------------------------------------------------------------------
__global__ void avgpool_kernel(const half_t* __restrict__ in, half_t* __restrict__ out) {
    int i = blockIdx.x * blockDim.x + threadIdx.x;
    if (i < 128 * 512) {
        float s = 0.0f;
#pragma unroll
        for (int k = 0; k < 16; ++k) s += (float)in[(size_t)i * 16 + k];
        out[i] = (half_t)(s * 0.0625f);
    }
}

__global__ void fc_bias_kernel(const float* __restrict__ x, const float* __restrict__ b,
                               float* __restrict__ out) {
    int i = blockIdx.x * blockDim.x + threadIdx.x;
    if (i < 128 * 512) out[i] = x[i] + b[i & 511];
}

// ---------------------------------------------------------------------------
// host driver
// ---------------------------------------------------------------------------
extern "C" void kernel_launch(void* const* d_in, const int* in_sizes, int n_in,
                              void* d_out, int out_size, void* d_ws, size_t ws_size,
                              hipStream_t stream) {
    (void)in_sizes; (void)n_in; (void)out_size; (void)ws_size;

    const float* input_obs = (const float*)d_in[0];
    const float* seg       = (const float*)d_in[1];
    // params flattened in alphabetical pytree-key order
    const float* bn1_b   = (const float*)d_in[2];
    const float* bn1_g   = (const float*)d_in[3];
    const float* conv1_w = (const float*)d_in[4];
    const float* fc_b    = (const float*)d_in[5];
    const float* fc_w    = (const float*)d_in[6];

    struct BlockP { const float *b1, *b2, *db, *dg, *dw, *g1, *g2, *w1, *w2; };
    BlockP bp[4][2];
    int pi = 7;
    for (int i = 0; i < 4; ++i)
        for (int j = 0; j < 2; ++j) {
            bool down = (j == 0 && i > 0);
            BlockP p{};
            p.b1 = (const float*)d_in[pi++];
            p.b2 = (const float*)d_in[pi++];
            if (down) {
                p.db = (const float*)d_in[pi++];
                p.dg = (const float*)d_in[pi++];
                p.dw = (const float*)d_in[pi++];
            }
            p.g1 = (const float*)d_in[pi++];
            p.g2 = (const float*)d_in[pi++];
            p.w1 = (const float*)d_in[pi++];
            p.w2 = (const float*)d_in[pi++];
            bp[i][j] = p;
        }

    // ---- workspace sub-allocation ----
    char* ws = (char*)d_ws;
    size_t off = 0;
    auto alloc = [&](size_t bytes) -> void* {
        void* p = ws + off;
        off += (bytes + 255) & ~(size_t)255;
        return p;
    };
    half_t* actA   = (half_t*)alloc((size_t)12582912 * 2);   // x / block io
    half_t* actB   = (half_t*)alloc((size_t)33554432 * 2);   // conv1 bn-relu / mids
    float*  convf  = (float*) alloc((size_t)33554432 * 4);   // conv f32 output
    float*  residf = (float*) alloc((size_t)4194304  * 4);   // downsample residual
    float*  meanA  = (float*) alloc(512 * 4);
    float*  rstdA  = (float*) alloc(512 * 4);
    float*  meanB  = (float*) alloc(512 * 4);
    float*  rstdB  = (float*) alloc(512 * 4);
    float*  anynz  = (float*) alloc(128 * 4);
    half_t* warena = (half_t*)alloc((size_t)12500000 * 2);

    // ---- weight repack to f16 padded-K rows ----
    size_t woff = 0;
    auto wrepack = [&](const float* src, int Cout, int Ktot) -> half_t* {
        int Kpad = (Ktot + 31) & ~31;
        half_t* dst = warena + woff;
        size_t n = (size_t)Cout * Kpad;
        woff += (n + 127) & ~(size_t)127;
        repack_w_kernel<<<(int)((n + 255) / 256), 256, 0, stream>>>(src, dst, Cout, Ktot, Kpad);
        return dst;
    };
    half_t* w_conv1 = wrepack(conv1_w, 64, 6 * 49);
    half_t* w_fc    = wrepack(fc_w, 512, 512);
    const int Cs[4] = {64, 128, 256, 512};
    half_t* wb1[4][2]; half_t* wb2[4][2]; half_t* wbd[4][2];
    {
        int cin = 64;
        for (int i = 0; i < 4; ++i) {
            int C = Cs[i];
            for (int j = 0; j < 2; ++j) {
                int ci = (j == 0) ? cin : C;
                wb1[i][j] = wrepack(bp[i][j].w1, C, ci * 9);
                wb2[i][j] = wrepack(bp[i][j].w2, C, C * 9);
                wbd[i][j] = (j == 0 && i > 0) ? wrepack(bp[i][j].dw, C, ci) : nullptr;
            }
            cin = C;
        }
    }

    // ---- launch helpers ----
    auto launch_conv = [&](int khw, int s, const half_t* in, const half_t* wt, float* out,
                           int Nimg, int Cin, int H, int W, int Cout) {
        int pad = (khw == 7) ? 3 : (khw == 3 ? 1 : 0);
        int OH = (H + 2 * pad - khw) / s + 1;
        int OW = (W + 2 * pad - khw) / s + 1;
        int Npix = Nimg * OH * OW;
        int blocks = (Cout / 16) * ((Npix + 511) / 512);
        if (khw == 7)
            conv_wmma_kernel<7, 2, 3><<<blocks, 256, 0, stream>>>(in, wt, out, Nimg, Cin, H, W, Cout, OH, OW);
        else if (khw == 3 && s == 1)
            conv_wmma_kernel<3, 1, 1><<<blocks, 256, 0, stream>>>(in, wt, out, Nimg, Cin, H, W, Cout, OH, OW);
        else if (khw == 3 && s == 2)
            conv_wmma_kernel<3, 2, 1><<<blocks, 256, 0, stream>>>(in, wt, out, Nimg, Cin, H, W, Cout, OH, OW);
        else if (khw == 1 && s == 2)
            conv_wmma_kernel<1, 2, 0><<<blocks, 256, 0, stream>>>(in, wt, out, Nimg, Cin, H, W, Cout, OH, OW);
        else
            conv_wmma_kernel<1, 1, 0><<<blocks, 256, 0, stream>>>(in, wt, out, Nimg, Cin, H, W, Cout, OH, OW);
    };
    auto bn_stats = [&](const float* x, int Nimg, int C, int HW, float* m, float* r) {
        bn_stats_kernel<<<C, 256, 0, stream>>>(x, Nimg, C, HW, m, r, 1e-5f);
    };
    auto bn_apply = [&](const float* x, int Nimg, int C, int HW, const float* m,
                        const float* r, const float* g, const float* b,
                        const float* rF, const half_t* rH, int relu,
                        half_t* oH, float* oF) {
        int gx = (HW + 255) / 256;
        if (gx > 8) gx = 8;
        dim3 grid(gx, Nimg * C);
        bn_apply_kernel<<<grid, 256, 0, stream>>>(x, m, r, g, b, C, HW, rF, rH, relu, oH, oF);
    };

    float* out_img   = (float*)d_out;                   // (16,8,512)
    float* out_early = out_img + 16 * 8 * 512;          // (16,8,64,32,32)

    // ---- preprocess ----
    anynz_kernel<<<128, 256, 0, stream>>>(seg, anynz);
    preprocess_kernel<<<4096, 256, 0, stream>>>(input_obs, seg, anynz, actA);

    // ---- stem: conv1 -> bn -> relu -> maxpool ----
    launch_conv(7, 2, actA, w_conv1, convf, 128, 6, 128, 128, 64);     // (128,64,64,64)
    bn_stats(convf, 128, 64, 64 * 64, meanA, rstdA);
    bn_apply(convf, 128, 64, 64 * 64, meanA, rstdA, bn1_g, bn1_b,
             nullptr, nullptr, 1, actB, nullptr);
    maxpool_kernel<<<(128 * 64 * 32 * 32 + 255) / 256, 256, 0, stream>>>(actB, actA, out_early);

    // ---- residual layers ----
    int cin = 64, HH = 32;
    for (int i = 0; i < 4; ++i) {
        int C = Cs[i];
        for (int j = 0; j < 2; ++j) {
            int s = (j == 0 && i > 0) ? 2 : 1;
            bool down = (j == 0 && i > 0);
            int OHs = (HH + 2 - 3) / s + 1;            // 3x3 pad1 output
            int HWo = OHs * OHs;

            launch_conv(3, s, actA, wb1[i][j], convf, 128, cin, HH, HH, C);
            bn_stats(convf, 128, C, HWo, meanA, rstdA);
            bn_apply(convf, 128, C, HWo, meanA, rstdA, bp[i][j].g1, bp[i][j].b1,
                     nullptr, nullptr, 1, actB, nullptr);

            launch_conv(3, 1, actB, wb2[i][j], convf, 128, C, OHs, OHs, C);
            bn_stats(convf, 128, C, HWo, meanA, rstdA);

            if (down) {
                launch_conv(1, s, actA, wbd[i][j], residf, 128, cin, HH, HH, C);
                bn_stats(residf, 128, C, HWo, meanB, rstdB);
                bn_apply(residf, 128, C, HWo, meanB, rstdB, bp[i][j].dg, bp[i][j].db,
                         nullptr, nullptr, 0, nullptr, residf);   // in-place BN
                bn_apply(convf, 128, C, HWo, meanA, rstdA, bp[i][j].g2, bp[i][j].b2,
                         residf, nullptr, 1, actA, nullptr);
            } else {
                bn_apply(convf, 128, C, HWo, meanA, rstdA, bp[i][j].g2, bp[i][j].b2,
                         nullptr, actA, 1, actA, nullptr);        // f16 residual, in-place
            }
            cin = C;
            HH = OHs;
        }
    }

    // ---- head: avgpool -> FC (1x1 WMMA conv) + bias ----
    avgpool_kernel<<<(65536 + 255) / 256, 256, 0, stream>>>(actA, actB);
    launch_conv(1, 1, actB, w_fc, convf, 128, 512, 1, 1, 512);
    fc_bias_kernel<<<(65536 + 255) / 256, 256, 0, stream>>>(convf, fc_b, out_img);
}
